// RNNAttnDecoder_54537494724950
// MI455X (gfx1250) — matrix operand order repaired and checked by reading, not compiled, not run on hardware
//
#include <hip/hip_runtime.h>

// RNN attention decoder for MI455X (gfx1250, wave32, WMMA).
// B=2048, T=48 (47 steps), S=128, H=256, V=96, L=2.
// One workgroup = 16 batch rows for all 47 steps; 16 waves (512 threads),
// each wave owns one 16-column N-tile -> only 6 live accumulators per lane
// (no spills). One-hot embedding handled by WMMA against W_ih0[:, :96].

#define HB 256
#define VB 96
#define SB 128
#define BB 2048
#define TT 48
#define NSTEP 47

// padded LDS strides (elements) - keep 16B alignment, avoid bank conflicts
#define AST 264   // 256 + 8
#define WST 520   // 512 + 8
#define OST 104   //  96 + 8

typedef __attribute__((ext_vector_type(16))) __bf16 v16bf;
typedef __attribute__((ext_vector_type(8)))  __bf16 v8bf;
typedef __attribute__((ext_vector_type(8)))  float  v8f;

__device__ __forceinline__ v8f wmma_bf16(v16bf a, v16bf b, v8f c) {
  return __builtin_amdgcn_wmma_f32_16x16x32_bf16(false, a, false, b, (short)0, c,
                                                 false, false);
}

// A-matrix 16x32 bf16 fragment from row-major [16][stride] LDS buffer.
__device__ __forceinline__ v16bf load_a_frag(const __bf16* base, int stride, int k0) {
  const int lane = threadIdx.x & 31;
  const int m = lane & 15, kh = lane >> 4;
  const __bf16* p = base + m * stride + k0 + kh * 16;
  v8bf lo = *(const v8bf*)p;
  v8bf hi = *(const v8bf*)(p + 8);
  v16bf r;
  ((v8bf*)&r)[0] = lo;
  ((v8bf*)&r)[1] = hi;
  return r;
}

// B-matrix 32x16 bf16 fragment from row-major weight W[n][K] starting at row n0.
__device__ __forceinline__ v16bf load_b_frag(const __bf16* w_n0, int K, int k0) {
  const int lane = threadIdx.x & 31;
  const int n = lane & 15, kh = lane >> 4;
  return *(const v16bf*)(w_n0 + (size_t)n * K + k0 + kh * 16);
}

__device__ __forceinline__ float sigm(float x) { return 1.0f / (1.0f + __expf(-x)); }
__device__ __forceinline__ float tanh_fast(float x) {
  float t = __expf(-2.0f * fabsf(x));
  float r = (1.0f - t) / (1.0f + t);
  return copysignf(r, x);
}

// ---------------- weight conversion (fp32 -> bf16 in workspace) ----------------
// layout (elements):
//  [0)        Wih0h [768][256]  = W_ih0[:, 96:352]
//  [196608)   Whh0  [768][256]
//  [393216)   Wih1  [768][256]
//  [589824)   Whh1  [768][256]
//  [786432)   Wc    [256][512]
//  [917504)   Ws    [ 96][256]
//  [942080)   Wih0v [768][ 96]  = W_ih0[:, :96]   (one-hot / embedding part)
//  total 1,015,808 elems
__global__ void cvt_weights(const float* __restrict__ wih0, const float* __restrict__ whh0,
                            const float* __restrict__ wih1, const float* __restrict__ whh1,
                            const float* __restrict__ wcw,  const float* __restrict__ wsw,
                            __bf16* __restrict__ o) {
  const int TOT = 1015808;
  for (int i = blockIdx.x * blockDim.x + threadIdx.x; i < TOT;
       i += gridDim.x * blockDim.x) {
    float v;
    if (i < 196608)      { int n = i >> 8, k = i & 255; v = wih0[n * 352 + 96 + k]; }
    else if (i < 393216) v = whh0[i - 196608];
    else if (i < 589824) v = wih1[i - 393216];
    else if (i < 786432) v = whh1[i - 589824];
    else if (i < 917504) v = wcw[i - 786432];
    else if (i < 942080) v = wsw[i - 917504];
    else { int j = i - 942080; int n = j / 96, k = j - n * 96; v = wih0[n * 352 + k]; }
    o[i] = (__bf16)v;
  }
}

__global__ void cvt_enc(const float* __restrict__ in, __bf16* __restrict__ o, int n8) {
  for (int i = blockIdx.x * blockDim.x + threadIdx.x; i < n8;
       i += gridDim.x * blockDim.x) {
    float4 a = ((const float4*)in)[i * 2];
    float4 b = ((const float4*)in)[i * 2 + 1];
    v8bf r;
    r[0] = (__bf16)a.x; r[1] = (__bf16)a.y; r[2] = (__bf16)a.z; r[3] = (__bf16)a.w;
    r[4] = (__bf16)b.x; r[5] = (__bf16)b.y; r[6] = (__bf16)b.z; r[7] = (__bf16)b.w;
    ((v8bf*)o)[i] = r;
  }
}

// ---------------- main decoder ----------------
__global__ __launch_bounds__(512, 1)
void rnn_attn_decoder(const int* __restrict__ y,
                      const float* __restrict__ enc,
                      const __bf16* __restrict__ encb,   // bf16 copy or nullptr
                      const float* __restrict__ ehid,
                      const __bf16* __restrict__ Wih0h, const __bf16* __restrict__ Whh0,
                      const __bf16* __restrict__ Wih1,  const __bf16* __restrict__ Whh1,
                      const __bf16* __restrict__ Wc,    const __bf16* __restrict__ Ws,
                      const __bf16* __restrict__ Wih0v,
                      const float* __restrict__ bih0, const float* __restrict__ bhh0,
                      const float* __restrict__ bih1, const float* __restrict__ bhh1,
                      const float* __restrict__ bwc,  const float* __restrict__ bws,
                      float* __restrict__ out) {
  __shared__ __bf16 c0b[16 * AST];    // bf16 GRU0 state (A operand)
  __shared__ __bf16 c1b[16 * AST];    // bf16 GRU1 state (A operand)
  __shared__ __bf16 albt[16 * AST];   // last_ht (A for GRU0 gi, A for logits)
  __shared__ __bf16 awc[16 * WST];    // [out | ctx] (A for wc GEMM)
  __shared__ __bf16 oneh[16 * OST];   // one-hot(y_t) (A for embedding GEMM)
  __shared__ float  alpha_s[16 * SB];
  __shared__ int    sy[16];

  const int tid   = threadIdx.x;
  const int lane  = tid & 31;
  const int wave  = tid >> 5;          // 0..15, owns N-columns [wave*16, wave*16+16)
  const int nl    = lane & 15;
  const int mBase = (lane >> 4) * 8;
  const int b0    = blockIdx.x * 16;
  const int nc    = wave * 16 + nl;    // this lane's output column within H

  for (int i = tid; i < 16 * AST; i += 512) albt[i] = (__bf16)0.0f;  // last_ht = 0
  for (int i = tid; i < 16 * OST; i += 512) oneh[i] = (__bf16)0.0f;
  if (tid < 16) sy[tid] = 0;

  // register-resident GRU states (this wave's 16 columns)
  float c0r[8], c1r[8];
#pragma unroll
  for (int g = 0; g < 8; ++g) {
    const int m = mBase + g;
    float a0 = ehid[(size_t)(b0 + m) * HB + nc];
    float a1 = ehid[(size_t)BB * HB + (size_t)(b0 + m) * HB + nc];
    c0r[g] = a0; c1r[g] = a1;
    c0b[m * AST + nc] = (__bf16)a0;
    c1b[m * AST + nc] = (__bf16)a1;
  }
  __syncthreads();

  for (int t = 0; t < NSTEP; ++t) {
    // incremental one-hot update (thread tid owns row tid)
    if (tid < 16) {
      oneh[tid * OST + sy[tid]] = (__bf16)0.0f;   // clear previous position
      int v = y[(b0 + tid) * TT + t];
      sy[tid] = v;
      oneh[tid * OST + v] = (__bf16)1.0f;
    }
    __syncthreads();

    // ================= GRU layer 0 =================
    {
      v8f gi[3], gh[3];
#pragma unroll
      for (int gate = 0; gate < 3; ++gate) {
        float bi = bih0[gate * 256 + nc];
        float bh = bhh0[gate * 256 + nc];
#pragma unroll
        for (int g = 0; g < 8; ++g) { gi[gate][g] = bi; gh[gate][g] = bh; }
      }
      // one-hot (embedding) part: K = 96
#pragma unroll 1
      for (int k0 = 0; k0 < 96; k0 += 32) {
        v16bf aV = load_a_frag(oneh, OST, k0);
#pragma unroll
        for (int gate = 0; gate < 3; ++gate) {
          const size_t r0 = (size_t)(gate * 256 + wave * 16);
          gi[gate] = wmma_bf16(aV, load_b_frag(Wih0v + r0 * 96, 96, k0), gi[gate]);
        }
      }
      // hidden part: K = 256
#pragma unroll 1
      for (int k0 = 0; k0 < 256; k0 += 32) {
        v16bf aX = load_a_frag(albt, AST, k0);
        v16bf aH = load_a_frag(c0b, AST, k0);
#pragma unroll
        for (int gate = 0; gate < 3; ++gate) {
          const size_t r0 = (size_t)(gate * 256 + wave * 16);
          gi[gate] = wmma_bf16(aX, load_b_frag(Wih0h + r0 * 256, 256, k0), gi[gate]);
          gh[gate] = wmma_bf16(aH, load_b_frag(Whh0  + r0 * 256, 256, k0), gh[gate]);
        }
      }
#pragma unroll
      for (int g = 0; g < 8; ++g) {
        float r  = sigm(gi[0][g] + gh[0][g]);
        float z  = sigm(gi[1][g] + gh[1][g]);
        float nn = tanh_fast(gi[2][g] + r * gh[2][g]);
        c0r[g] = (1.0f - z) * nn + z * c0r[g];
      }
    }
    __syncthreads();  // all waves done reading c0b as A
#pragma unroll
    for (int g = 0; g < 8; ++g) c0b[(mBase + g) * AST + nc] = (__bf16)c0r[g];
    __syncthreads();

    // ================= GRU layer 1 =================
    {
      v8f gi[3], gh[3];
#pragma unroll
      for (int gate = 0; gate < 3; ++gate) {
        float bi = bih1[gate * 256 + nc];
        float bh = bhh1[gate * 256 + nc];
#pragma unroll
        for (int g = 0; g < 8; ++g) { gi[gate][g] = bi; gh[gate][g] = bh; }
      }
#pragma unroll 1
      for (int k0 = 0; k0 < 256; k0 += 32) {
        v16bf aX = load_a_frag(c0b, AST, k0);
        v16bf aH = load_a_frag(c1b, AST, k0);
#pragma unroll
        for (int gate = 0; gate < 3; ++gate) {
          const size_t r0 = (size_t)(gate * 256 + wave * 16);
          gi[gate] = wmma_bf16(aX, load_b_frag(Wih1 + r0 * 256, 256, k0), gi[gate]);
          gh[gate] = wmma_bf16(aH, load_b_frag(Whh1 + r0 * 256, 256, k0), gh[gate]);
        }
      }
#pragma unroll
      for (int g = 0; g < 8; ++g) {
        float r  = sigm(gi[0][g] + gh[0][g]);
        float z  = sigm(gi[1][g] + gh[1][g]);
        float nn = tanh_fast(gi[2][g] + r * gh[2][g]);
        c1r[g] = (1.0f - z) * nn + z * c1r[g];
      }
    }
    __syncthreads();  // all waves done reading c1b as A
#pragma unroll
    for (int g = 0; g < 8; ++g) {
      __bf16 hb = (__bf16)c1r[g];
      c1b[(mBase + g) * AST + nc] = hb;   // next step's gh1 A + attention "out"
      awc[(mBase + g) * WST + nc] = hb;   // first half of [out|ctx]
    }
    __syncthreads();

    // ================= attention: energy + softmax =================
    // wave w owns batch row m = w; 32 lanes cover S=128 (4 s-values per lane)
    {
      const int m = wave;
      float e[4];
#pragma unroll
      for (int si = 0; si < 4; ++si) e[si] = 0.0f;

      if (encb) {
        const __bf16* eb = encb + (size_t)(b0 + m) * (SB * HB);
        for (int h0 = 0; h0 < HB; h0 += 8) {
          v8bf ov = *(const v8bf*)(c1b + m * AST + h0);
          float of[8];
#pragma unroll
          for (int i = 0; i < 8; ++i) of[i] = (float)ov[i];
#pragma unroll
          for (int si = 0; si < 4; ++si) {
            v8bf ex = *(const v8bf*)(eb + (size_t)(lane * 4 + si) * HB + h0);
            float s = 0.0f;
#pragma unroll
            for (int i = 0; i < 8; ++i) s += of[i] * (float)ex[i];
            e[si] += s;
          }
        }
      } else {
        const float* eb = enc + (size_t)(b0 + m) * (SB * HB);
        for (int h0 = 0; h0 < HB; h0 += 8) {
          v8bf ov = *(const v8bf*)(c1b + m * AST + h0);
          float of[8];
#pragma unroll
          for (int i = 0; i < 8; ++i) of[i] = (float)ov[i];
#pragma unroll
          for (int si = 0; si < 4; ++si) {
            const float* row = eb + (size_t)(lane * 4 + si) * HB + h0;
            float4 x0 = *(const float4*)row;
            float4 x1 = *(const float4*)(row + 4);
            e[si] += of[0] * x0.x + of[1] * x0.y + of[2] * x0.z + of[3] * x0.w +
                     of[4] * x1.x + of[5] * x1.y + of[6] * x1.z + of[7] * x1.w;
          }
        }
      }
      // softmax over 128 values across the full wave
      float mx = e[0];
#pragma unroll
      for (int si = 1; si < 4; ++si) mx = fmaxf(mx, e[si]);
#pragma unroll
      for (int off = 1; off < 32; off <<= 1) mx = fmaxf(mx, __shfl_xor(mx, off, 32));
      float sum = 0.0f;
#pragma unroll
      for (int si = 0; si < 4; ++si) { e[si] = __expf(e[si] - mx); sum += e[si]; }
#pragma unroll
      for (int off = 1; off < 32; off <<= 1) sum += __shfl_xor(sum, off, 32);
      float inv = 1.0f / sum;
#pragma unroll
      for (int si = 0; si < 4; ++si) alpha_s[m * SB + lane * 4 + si] = e[si] * inv;
    }
    __syncthreads();

    // ================= attention: context =================
    // wave w owns row m = w; 32 lanes cover H=256 (8 h-values per lane)
    {
      const int m   = wave;
      const int h0c = lane * 8;
      float cx[8];
#pragma unroll
      for (int i = 0; i < 8; ++i) cx[i] = 0.0f;

      if (encb) {
        const __bf16* eb = encb + (size_t)(b0 + m) * (SB * HB) + h0c;
        for (int s = 0; s < SB; ++s) {
          float a = alpha_s[m * SB + s];
          v8bf x = *(const v8bf*)(eb + (size_t)s * HB);
#pragma unroll
          for (int i = 0; i < 8; ++i) cx[i] += a * (float)x[i];
        }
      } else {
        const float* eb = enc + (size_t)(b0 + m) * (SB * HB) + h0c;
        for (int s = 0; s < SB; ++s) {
          float a = alpha_s[m * SB + s];
          const float* row = eb + (size_t)s * HB;
          float4 r0 = *(const float4*)row;
          float4 r1 = *(const float4*)(row + 4);
          cx[0] += a * r0.x; cx[1] += a * r0.y; cx[2] += a * r0.z; cx[3] += a * r0.w;
          cx[4] += a * r1.x; cx[5] += a * r1.y; cx[6] += a * r1.z; cx[7] += a * r1.w;
        }
      }
#pragma unroll
      for (int i = 0; i < 8; ++i) awc[m * WST + 256 + h0c + i] = (__bf16)cx[i];
    }
    __syncthreads();

    // ================= ht = tanh([out|ctx] @ wc_W.T + wc_b) =================
    {
      v8f ha;
      float bw = bwc[nc];
#pragma unroll
      for (int g = 0; g < 8; ++g) ha[g] = bw;
#pragma unroll 1
      for (int k0 = 0; k0 < 512; k0 += 32) {
        v16bf a = load_a_frag(awc, WST, k0);
        ha = wmma_bf16(a, load_b_frag(Wc + (size_t)(wave * 16) * 512, 512, k0), ha);
      }
#pragma unroll
      for (int g = 0; g < 8; ++g)
        albt[(mBase + g) * AST + nc] = (__bf16)tanh_fast(ha[g]);  // next last_ht / logits A
    }
    __syncthreads();

    // ================= logits = ht @ ws_W.T + ws_b =================
    if (wave < 6) {  // N = 96 -> 6 tiles of 16
      const int n0 = wave * 16;
      const int n  = n0 + nl;
      v8f la;
      float bv = bws[n];
#pragma unroll
      for (int g = 0; g < 8; ++g) la[g] = bv;
#pragma unroll 1
      for (int k0 = 0; k0 < 256; k0 += 32) {
        v16bf a = load_a_frag(albt, AST, k0);
        la = wmma_bf16(a, load_b_frag(Ws + (size_t)n0 * 256, 256, k0), la);
      }
#pragma unroll
      for (int g = 0; g < 8; ++g) {
        const int m = mBase + g;
        out[((size_t)(b0 + m) * NSTEP + t) * VB + n] = la[g];
      }
    }
    __syncthreads();
  }
}

extern "C" void kernel_launch(void* const* d_in, const int* in_sizes, int n_in,
                              void* d_out, int out_size, void* d_ws, size_t ws_size,
                              hipStream_t stream) {
  (void)in_sizes; (void)n_in; (void)out_size;
  const int*   y    = (const int*)d_in[0];
  const float* enc  = (const float*)d_in[1];
  const float* ehid = (const float*)d_in[2];
  // d_in[3] = is_training (always teacher forcing per reference)
  const float* wih0 = (const float*)d_in[4];
  const float* whh0 = (const float*)d_in[5];
  const float* bih0 = (const float*)d_in[6];
  const float* bhh0 = (const float*)d_in[7];
  const float* wih1 = (const float*)d_in[8];
  const float* whh1 = (const float*)d_in[9];
  const float* bih1 = (const float*)d_in[10];
  const float* bhh1 = (const float*)d_in[11];
  const float* wcw  = (const float*)d_in[12];
  const float* wcb  = (const float*)d_in[13];
  const float* wsw  = (const float*)d_in[14];
  const float* wsb  = (const float*)d_in[15];
  float* out = (float*)d_out;

  __bf16* W = (__bf16*)d_ws;
  const size_t W_BYTES = 1015808ull * 2ull;
  if (ws_size < W_BYTES) return;  // cannot run without bf16 weight staging

  cvt_weights<<<512, 256, 0, stream>>>(wih0, whh0, wih1, whh1, wcw, wsw, W);

  // bf16 encoder copy (128 MB -> L2-resident) if workspace allows
  const size_t ENC_BYTE_OFF = 2097152;  // 2 MiB
  const size_t ENC_BYTES = (size_t)BB * SB * HB * 2ull;
  __bf16* encB = nullptr;
  if (ws_size >= ENC_BYTE_OFF + ENC_BYTES) {
    encB = W + (ENC_BYTE_OFF / 2);
    cvt_enc<<<2048, 256, 0, stream>>>(enc, encB, (BB * SB * HB) / 8);
  }

  rnn_attn_decoder<<<BB / 16, 512, 0, stream>>>(
      y, enc, encB, ehid,
      W + 0, W + 196608, W + 393216, W + 589824,   // Wih0h, Whh0, Wih1, Whh1
      W + 786432, W + 917504, W + 942080,          // Wc, Ws, Wih0v
      bih0, bhh0, bih1, bhh1, wcb, wsb, out);
}
